// QuantumNN_57732950392970
// MI455X (gfx1250) — compile-verified, matrix operand order checked by reading, not run on hardware
//
#include <hip/hip_runtime.h>
#include <hip/hip_bf16.h>

// ---------------------------------------------------------------------------
// QuantumNN fused pipeline for MI455X (gfx1250, wave32, WMMA, async-to-LDS)
//   k0: We1T = f16 transpose of We1 (once, 0.4 MB, enables async f16 staging)
//   k1: h  = relu(x @ We1 + be1)  f16 [65536,256]; B panel double-buffered via
//       global_load_async_to_lds_b128, A prefetched in regs -> LDS ping-pong
//   k2: enc = tanh(h @ We2 + be2) -> quantum circuit (per-lane, registers)
//       -> feat -> relu(feat @ W1 + b1) @ W2 + b2 -> out f32 [65536,10]
// All GEMMs use v_wmma_f32_16x16x32_f16 with f32 accumulation.
// ---------------------------------------------------------------------------

typedef __attribute__((ext_vector_type(16))) _Float16 v16h;
typedef __attribute__((ext_vector_type(8)))  _Float16 v8h;
typedef __attribute__((ext_vector_type(4)))  _Float16 v4h;
typedef __attribute__((ext_vector_type(8)))  float    v8f;

#define B_TOTAL 65536
#define K1 784
#define N1 256
#define KPAD 832          // 13 * 64, zero-padded K for We1T

// ---- fragment loader: 16-bit A/B 16x32 fragment (CDNA5 ISA 7.12.2 layout)
// lanes 0-15 : row/col = lane,    K = k0+0..7   (VGPR0-3),  k0+16..23 (VGPR4-7)
// lanes 16-31: row/col = lane-16, K = k0+8..15  (VGPR0-3),  k0+24..31 (VGPR4-7)
__device__ __forceinline__ v16h load_frag16(const _Float16* base, int row_stride,
                                            int rbase, int k0) {
  const int lane = threadIdx.x & 31;
  const int r = rbase + (lane & 15);
  const int k = k0 + ((lane >> 4) << 3);
  const _Float16* p = base + r * row_stride + k;
  union { v16h v; v8h q[2]; } u;
  u.q[0] = *reinterpret_cast<const v8h*>(p);        // 16B -> ds/global _b128
  u.q[1] = *reinterpret_cast<const v8h*>(p + 16);
  return u.v;
}

__device__ __forceinline__ v8f wmma_f16(v16h a, v16h b, v8f c) {
  return __builtin_amdgcn_wmma_f32_16x16x32_f16(false, a, false, b,
                                                (short)0, c, false, false);
}

// ===========================================================================
// Kernel 0: We1T[n][k] = (f16)We1[k][n], K padded 784 -> 832 with zeros
// ===========================================================================
__global__ __launch_bounds__(256)
void k0_transpose_we1(const float* __restrict__ We1, _Float16* __restrict__ We1T) {
  const int n = blockIdx.x;                  // 0..255
  for (int k = threadIdx.x; k < KPAD; k += 256)
    We1T[n * KPAD + k] = (k < K1) ? (_Float16)We1[k * N1 + n] : (_Float16)0.f;
}

// ===========================================================================
// Kernel 1: h = relu(x @ We1 + be1), f16 output
// WG: 256 thr = 8 waves x 16 rows = 128 rows. K chunked by 64.
// B panel: async-to-LDS double buffer. A panel: reg prefetch -> LDS ping-pong.
// ===========================================================================
#define MB1 128
#define KC 64
#define AS_STRIDE 72      // halves; multiple of 8 (16B frags), spreads LDS banks
#define BT_STRIDE 72
#define A_CHUNK_H (MB1 * AS_STRIDE)          // halves per A buffer
#define B_CHUNK_H (N1 * BT_STRIDE)           // halves per B buffer
#define K1_SMEM_BYTES ((2 * A_CHUNK_H + 2 * B_CHUNK_H) * 2)   // 110592 B

__global__ __launch_bounds__(256)
void k1_gemm_relu(const float* __restrict__ x, const _Float16* __restrict__ We1T,
                  const float* __restrict__ be1, _Float16* __restrict__ h) {
  extern __shared__ char smem1[];
  _Float16* Abuf = (_Float16*)smem1;                          // 2 x [128][72]
  _Float16* Bbuf = (_Float16*)(smem1 + 2 * A_CHUNK_H * 2);    // 2 x [256][72]

  const int tid  = threadIdx.x;
  const int lane = tid & 31;
  const int wv   = tid >> 5;
  const int row0 = blockIdx.x * MB1;
  const int nchunks = (K1 + KC - 1) / KC;                     // 13

  // byte offset of Bbuf inside this wave's LDS window (dynamic LDS follows static)
  const unsigned ldsB0 = __builtin_amdgcn_groupstaticsize() + 2u * A_CHUNK_H * 2u;
  const unsigned long long we1t_base = (unsigned long long)(uintptr_t)We1T;

  const v8f vzero = {};
  v8f acc[16];
  #pragma unroll
  for (int i = 0; i < 16; ++i) acc[i] = vzero;

  float4 areg[8];

  // -- async stage of one B chunk (f16, pre-transposed) into buffer (c&1) --
  auto issue_async_B = [&](int c) {
    const int kbase = c * KC;
    #pragma unroll
    for (int j = 0; j < 8; ++j) {               // 2048 x 16B segs / 256 thr
      int flat = j * 256 + tid;
      int n = flat >> 3;
      int s = flat & 7;
      unsigned lds = ldsB0 + (unsigned)(c & 1) * (B_CHUNK_H * 2)
                   + (unsigned)(n * BT_STRIDE + s * 8) * 2;
      unsigned goff = (unsigned)(n * KPAD + kbase + s * 8) * 2;
      asm volatile("global_load_async_to_lds_b128 %0, %1, %2 offset:0"
                   :: "v"(lds), "v"(goff), "s"(we1t_base) : "memory");
    }
  };
  // -- prefetch one A chunk (f32) into registers --
  auto load_A_regs = [&](int c) {
    const int kbase = c * KC;
    #pragma unroll
    for (int j = 0; j < 8; ++j) {               // 2048 x float4 / 256 thr
      int flat = j * 256 + tid;
      int r = flat >> 4;
      int s = flat & 15;
      int gk = kbase + s * 4;
      float4 v = make_float4(0.f, 0.f, 0.f, 0.f);
      if (gk < K1) v = *(const float4*)&x[(row0 + r) * K1 + gk];
      areg[j] = v;
    }
  };
  // -- convert + store A regs into buffer (c&1) --
  auto store_A_regs = [&](int c) {
    _Float16* Ad = Abuf + (c & 1) * A_CHUNK_H;
    #pragma unroll
    for (int j = 0; j < 8; ++j) {
      int flat = j * 256 + tid;
      int r = flat >> 4;
      int s = flat & 15;
      v4h hv;
      hv.x = (_Float16)areg[j].x; hv.y = (_Float16)areg[j].y;
      hv.z = (_Float16)areg[j].z; hv.w = (_Float16)areg[j].w;
      *(v4h*)&Ad[r * AS_STRIDE + s * 4] = hv;   // ds_store_b64, 8B aligned
    }
  };

  // prologue: fill buffer 0
  issue_async_B(0);
  load_A_regs(0);
  store_A_regs(0);
  asm volatile("s_wait_asynccnt 0x0" ::: "memory");
  __syncthreads();

  for (int c = 0; c < nchunks; ++c) {
    if (c + 1 < nchunks) {                      // overlap: next B via async DMA,
      issue_async_B(c + 1);                     // next A into registers
      load_A_regs(c + 1);
    }
    const _Float16* Ac = Abuf + (c & 1) * A_CHUNK_H;
    const _Float16* Bc = Bbuf + (c & 1) * B_CHUNK_H;
    #pragma unroll
    for (int kt = 0; kt < 2; ++kt) {
      v16h a = load_frag16(Ac, AS_STRIDE, wv * 16, kt * 32);
      v16h b = load_frag16(Bc, BT_STRIDE, 0, kt * 32);
      #pragma unroll
      for (int nt = 0; nt < 16; ++nt) {
        v16h bn = b;
        if (nt < 15) bn = load_frag16(Bc, BT_STRIDE, (nt + 1) * 16, kt * 32);
        acc[nt] = wmma_f16(a, b, acc[nt]);      // wmma overlaps next ds_load
        b = bn;
      }
    }
    if (c + 1 < nchunks) store_A_regs(c + 1);
    asm volatile("s_wait_asynccnt 0x0" ::: "memory");
    __syncthreads();
  }

  // epilogue: +bias, relu, f16 store. C layout: VGPR i -> M = i + 8*(lane>>4)
  const int mrow = row0 + wv * 16 + 8 * (lane >> 4);
  const int nc   = lane & 15;
  #pragma unroll
  for (int nt = 0; nt < 16; ++nt) {
    int N = nt * 16 + nc;
    float bias = be1[N];
    #pragma unroll
    for (int i = 0; i < 8; ++i) {
      float v = acc[nt][i] + bias;
      v = v > 0.f ? v : 0.f;
      h[(mrow + i) * N1 + N] = (_Float16)v;
    }
  }
}

// ===========================================================================
// Kernel 2: fused GEMM2 + tanh + quantum circuit + classifier
// WG: 256 thr = 8 waves x 32 rows = 256 rows/WG.  Dynamic LDS ~158KB (<320KB).
// ===========================================================================
#define BT2_STRIDE 264   // We2^T  [32][264]  (K=256)
#define W1T_STRIDE 40    // W1^T   [128][40]  (K=32)
#define W2T_STRIDE 136   // W2^T   [16][136]  (K=128, N padded 10->16)
#define ENC_STRIDE 36    // enc f32 [256][36]
#define FEAT_STRIDE 40   // feat f16 [256][40]
#define HID_STRIDE 136   // hid f16 [256][136]
#define K2_SMEM_BYTES (32*BT2_STRIDE*2 + 128*W1T_STRIDE*2 + 16*W2T_STRIDE*2 + \
                       256*ENC_STRIDE*4 + 256*FEAT_STRIDE*2 + 256*HID_STRIDE*2)

// ---- quantum gates on 16-dim complex state in registers (qubit 0 = MSB) ----
__device__ __forceinline__ void rx_gate(float* sr, float* si, int m, float c, float s) {
  #pragma unroll
  for (int i = 0; i < 16; ++i) {
    if (i & m) continue;
    int j = i | m;
    float ar = sr[i], ai = si[i], br = sr[j], bi = si[j];
    sr[i] = c * ar + s * bi;  si[i] = c * ai - s * br;   // a' = c a - i s b
    sr[j] = s * ai + c * br;  si[j] = -s * ar + c * bi;  // b' = -i s a + c b
  }
}
__device__ __forceinline__ void ry_gate(float* sr, float* si, int m, float c, float s) {
  #pragma unroll
  for (int i = 0; i < 16; ++i) {
    if (i & m) continue;
    int j = i | m;
    float ar = sr[i], ai = si[i], br = sr[j], bi = si[j];
    sr[i] = c * ar - s * br;  si[i] = c * ai - s * bi;   // a' = c a - s b
    sr[j] = s * ar + c * br;  si[j] = s * ai + c * bi;   // b' = s a + c b
  }
}
__device__ __forceinline__ void cnot_gate(float* sr, float* si, int mc, int mt) {
  #pragma unroll
  for (int i = 0; i < 16; ++i) {
    if ((i & mc) && !(i & mt)) {
      int j = i | mt;
      float tr = sr[i]; sr[i] = sr[j]; sr[j] = tr;
      float ti = si[i]; si[i] = si[j]; si[j] = ti;
    }
  }
}

__global__ __launch_bounds__(256)
void k2_fused(const _Float16* __restrict__ h,
              const float* __restrict__ We2, const float* __restrict__ be2,
              const float* __restrict__ W1,  const float* __restrict__ b1,
              const float* __restrict__ W2,  const float* __restrict__ b2,
              float* __restrict__ out) {
  extern __shared__ char smem[];
  _Float16* BT2   = (_Float16*)smem;
  _Float16* W1T   = BT2 + 32 * BT2_STRIDE;
  _Float16* W2T   = W1T + 128 * W1T_STRIDE;
  float*    ENCB  = (float*)(W2T + 16 * W2T_STRIDE);
  _Float16* FEATB = (_Float16*)(ENCB + 256 * ENC_STRIDE);
  _Float16* HIDB  = FEATB + 256 * FEAT_STRIDE;

  const int tid  = threadIdx.x;
  const int lane = tid & 31;
  const int wv   = tid >> 5;
  const int r0   = blockIdx.x * 256 + wv * 32;   // global row base for this wave
  const v8f vzero = {};

  // ---- stage transposed weights (once per WG) ----
  #pragma unroll 4
  for (int n = 0; n < 32; ++n)                   // BT2[n][k] = We2[k][n], k = tid
    BT2[n * BT2_STRIDE + tid] = (_Float16)We2[tid * 32 + n];
  if (tid < 128) {
    #pragma unroll 4
    for (int k = 0; k < 32; ++k)                 // W1T[n][k] = W1[k][n], n = tid
      W1T[tid * W1T_STRIDE + k] = (_Float16)W1[k * 128 + tid];
    #pragma unroll
    for (int n = 0; n < 16; ++n)                 // W2T[n][k] = W2[k][n], k = tid
      W2T[n * W2T_STRIDE + tid] = (n < 10) ? (_Float16)W2[tid * 10 + n]
                                           : (_Float16)0.f;
  }
  __syncthreads();

  const int m8 = 8 * (lane >> 4);
  const int nc = lane & 15;

  // ---- GEMM2: enc = tanh(h @ We2 + be2), M=32/wave, N=32, K=256 ----
  {
    v8f acc[2][2];
    #pragma unroll
    for (int mt = 0; mt < 2; ++mt)
      #pragma unroll
      for (int nt = 0; nt < 2; ++nt) acc[mt][nt] = vzero;

    for (int kt = 0; kt < 8; ++kt) {
      #pragma unroll
      for (int mt = 0; mt < 2; ++mt) {
        v16h a = load_frag16(h, N1, r0 + mt * 16, kt * 32);   // from global (L2)
        #pragma unroll
        for (int nt = 0; nt < 2; ++nt) {
          v16h b = load_frag16(BT2, BT2_STRIDE, nt * 16, kt * 32);
          acc[mt][nt] = wmma_f16(a, b, acc[mt][nt]);
        }
      }
    }
    #pragma unroll
    for (int mt = 0; mt < 2; ++mt)
      #pragma unroll
      for (int nt = 0; nt < 2; ++nt) {
        int N = nt * 16 + nc;
        float bias = be2[N];
        #pragma unroll
        for (int i = 0; i < 8; ++i) {
          int M = wv * 32 + mt * 16 + m8 + i;   // WG-local row
          ENCB[M * ENC_STRIDE + N] = tanhf(acc[mt][nt][i] + bias);
        }
      }
  }
  __syncthreads();

  // ---- quantum circuit: one full 16-amp state per lane (32 rows <-> 32 lanes)
  {
    const float* e = ENCB + (wv * 32 + lane) * ENC_STRIDE;
    float sr[16], si[16], nrm = 0.f;
    #pragma unroll
    for (int j = 0; j < 16; ++j) {
      sr[j] = e[j]; si[j] = e[16 + j];
      nrm += sr[j] * sr[j] + si[j] * si[j];
    }
    float inv = rsqrtf(nrm > 1e-30f ? nrm : 1e-30f);
    #pragma unroll
    for (int j = 0; j < 16; ++j) { sr[j] *= inv; si[j] *= inv; }

    const float c1 = cosf(0.05f), s1 = sinf(0.05f);   // RX(0.1)
    const float c2 = cosf(0.10f), s2 = sinf(0.10f);   // RY(0.2)
    const float c3 = cosf(0.15f), s3 = sinf(0.15f);   // RX(0.3)
    const float c4 = cosf(0.20f), s4 = sinf(0.20f);   // RY(0.4)

    #pragma unroll
    for (int q = 0; q < 4; ++q) {
      int m = 8 >> q;                                 // qubit 0 = MSB
      rx_gate(sr, si, m, c1, s1);
      ry_gate(sr, si, m, c2, s2);
    }
    cnot_gate(sr, si, 8, 4);
    cnot_gate(sr, si, 4, 2);
    cnot_gate(sr, si, 2, 1);
    #pragma unroll
    for (int q = 0; q < 4; ++q) {
      int m = 8 >> q;
      rx_gate(sr, si, m, c3, s3);
      ry_gate(sr, si, m, c4, s4);
    }
    _Float16* f = FEATB + (wv * 32 + lane) * FEAT_STRIDE;
    #pragma unroll
    for (int j = 0; j < 16; ++j) {
      f[j]      = (_Float16)sr[j];
      f[16 + j] = (_Float16)si[j];
    }
  }
  __syncthreads();

  // ---- GEMM3: hid = relu(feat @ W1 + b1), N=128, K=32 (single ktile) ----
  {
    v8f acc[2][8];
    #pragma unroll
    for (int mt = 0; mt < 2; ++mt)
      #pragma unroll
      for (int nt = 0; nt < 8; ++nt) acc[mt][nt] = vzero;
    #pragma unroll
    for (int mt = 0; mt < 2; ++mt) {
      v16h a = load_frag16(FEATB, FEAT_STRIDE, wv * 32 + mt * 16, 0);
      v16h b = load_frag16(W1T, W1T_STRIDE, 0, 0);
      #pragma unroll
      for (int nt = 0; nt < 8; ++nt) {
        v16h bn = b;
        if (nt < 7) bn = load_frag16(W1T, W1T_STRIDE, (nt + 1) * 16, 0);
        acc[mt][nt] = wmma_f16(a, b, acc[mt][nt]);
        b = bn;
      }
    }
    #pragma unroll
    for (int mt = 0; mt < 2; ++mt)
      #pragma unroll
      for (int nt = 0; nt < 8; ++nt) {
        int N = nt * 16 + nc;
        float bias = b1[N];
        #pragma unroll
        for (int i = 0; i < 8; ++i) {
          int M = wv * 32 + mt * 16 + m8 + i;
          float v = acc[mt][nt][i] + bias;
          HIDB[M * HID_STRIDE + N] = (_Float16)(v > 0.f ? v : 0.f);
        }
      }
  }
  __syncthreads();

  // ---- GEMM4: out = hid @ W2 + b2, N=10 (padded 16), K=128 ----
  {
    v8f acc[2] = {vzero, vzero};
    #pragma unroll
    for (int kt = 0; kt < 4; ++kt) {
      v16h b = load_frag16(W2T, W2T_STRIDE, 0, kt * 32);
      #pragma unroll
      for (int mt = 0; mt < 2; ++mt) {
        v16h a = load_frag16(HIDB, HID_STRIDE, wv * 32 + mt * 16, kt * 32);
        acc[mt] = wmma_f16(a, b, acc[mt]);
      }
    }
    if (nc < 10) {
      float bias = b2[nc];
      #pragma unroll
      for (int mt = 0; mt < 2; ++mt)
        #pragma unroll
        for (int i = 0; i < 8; ++i) {
          int row = r0 + mt * 16 + m8 + i;
          out[row * 10 + nc] = acc[mt][i] + bias;
        }
    }
  }
}

// ===========================================================================
extern "C" void kernel_launch(void* const* d_in, const int* in_sizes, int n_in,
                              void* d_out, int out_size, void* d_ws, size_t ws_size,
                              hipStream_t stream) {
  (void)in_sizes; (void)n_in; (void)out_size; (void)ws_size;
  const float* x   = (const float*)d_in[0];
  const float* We1 = (const float*)d_in[1];
  const float* be1 = (const float*)d_in[2];
  const float* We2 = (const float*)d_in[3];
  const float* be2 = (const float*)d_in[4];
  const float* W1  = (const float*)d_in[5];
  const float* b1  = (const float*)d_in[6];
  const float* W2  = (const float*)d_in[7];
  const float* b2  = (const float*)d_in[8];
  float* out = (float*)d_out;

  // workspace layout: h (33.5 MB) | We1T f16 [256][832] (0.43 MB)
  _Float16* h    = (_Float16*)d_ws;
  _Float16* We1T = (_Float16*)((char*)d_ws + (size_t)B_TOTAL * N1 * sizeof(_Float16));

  k0_transpose_we1<<<N1, 256, 0, stream>>>(We1, We1T);
  k1_gemm_relu<<<B_TOTAL / MB1, 256, K1_SMEM_BYTES, stream>>>(x, We1T, be1, h);
  k2_fused<<<B_TOTAL / 256, 256, K2_SMEM_BYTES, stream>>>(h, We2, be2, W1, b1,
                                                          W2, b2, out);
}
